// GRUNet_77799037600237
// MI455X (gfx1250) — matrix operand order repaired and checked by reading, not compiled, not run on hardware
//
#include <hip/hip_runtime.h>
#include <hip/hip_bf16.h>
#include <math.h>

// ---------------- types ----------------
typedef __bf16  v16bf __attribute__((ext_vector_type(16)));
typedef float   v8f   __attribute__((ext_vector_type(8)));
typedef float   fv4   __attribute__((ext_vector_type(4)));
typedef unsigned int uv4 __attribute__((ext_vector_type(4)));

union FragAB { v16bf v; uv4 q[2]; };          // 32 bytes = 16 bf16
union W8     { uv4 q; __bf16 h[8]; };          // 16 bytes = 8 bf16

#define HH   1024
#define G3H  3072
#define SEQ1 8192      // T*B
#define IN_D 512
#define BB   64
#define TT   128
#define NWG_REC 64

__device__ __forceinline__ float sigmoidf_(float x) { return 1.0f / (1.0f + expf(-x)); }

// ---------------- pack x (B,T,IN) f32 -> seq (T*B, IN) bf16 ----------------
__global__ __launch_bounds__(256) void pack_x_seq(const float* __restrict__ x,
                                                  __bf16* __restrict__ seq) {
    int id = blockIdx.x * 256 + threadIdx.x;          // < SEQ1*IN_D
    if (id >= SEQ1 * IN_D) return;
    int c = id & (IN_D - 1);
    int i = id >> 9;                                   // flat step t*B+b
    int t = i >> 6;                                    // /64
    int b = i & 63;
    seq[(size_t)id] = (__bf16)x[((size_t)(b * TT + t) << 9) + c];
}

// ---------------- f32 -> bf16 flat convert ----------------
__global__ __launch_bounds__(256) void cvt_bf16(const float* __restrict__ src,
                                                __bf16* __restrict__ dst, int n) {
    int id = blockIdx.x * 256 + threadIdx.x;
    if (id < n) dst[id] = (__bf16)src[id];
}

// ---------------- WMMA fragment loaders (ISA 7.12.2 layouts) ----------------
__device__ __forceinline__ FragAB loadFragA(const __bf16* aRow, int kb, int half) {
    FragAB f;
    // A 16x32 bf16: lane m=l15; VGPR0..3 -> K = kb+half*8+{0..7}; VGPR4..7 -> +16
    f.q[0] = *(const uv4*)(aRow + kb + half * 8);
    f.q[1] = *(const uv4*)(aRow + kb + 16 + half * 8);
    return f;
}
__device__ __forceinline__ FragAB loadFragB(const __bf16* wRow, int kb, int half) {
    FragAB f;
    // B 32x16: lane n=l15; VGPR0..3 -> K = kb+half*8+{0..7}; VGPR4..7 -> +16
    f.q[0] = *(const uv4*)(wRow + kb + half * 8);
    f.q[1] = *(const uv4*)(wRow + kb + 16 + half * 8);
    return f;
}

// ---------------- WMMA GEMM: C[M,N] = A[M,K] * W[N,K]^T + bias[N] ----------------
// A row-major bf16 (lda=K), W row-major bf16 (N rows of K), C row-major f32.
// Each wave computes a 16x64 tile; two-stage software pipeline so VMEM overlaps WMMA.
__global__ __launch_bounds__(256) void wmma_gemm_bias(const __bf16* __restrict__ A,
                                                      const __bf16* __restrict__ W,
                                                      const float*  __restrict__ bias,
                                                      float* __restrict__ C,
                                                      int M, int N, int K) {
    const int tid   = threadIdx.x;
    const int lane  = tid & 31;
    const int wid   = (blockIdx.x * 256 + tid) >> 5;   // global wave id
    const int numTM = M >> 4;
    const int numTN = N >> 6;
    if (wid >= numTM * numTN) return;

    const int tm = wid % numTM;            // consecutive waves share the N strip -> B reuse in L1
    const int tn = wid / numTM;
    const int mBase = tm << 4;
    const int nBase = tn << 6;

    const int half = lane >> 4;            // 0/1: K-half selector per ISA layout
    const int l15  = lane & 15;

    v8f acc[4];
    #pragma unroll
    for (int c = 0; c < 4; ++c)
        #pragma unroll
        for (int p = 0; p < 8; ++p) acc[c][p] = 0.0f;

    const __bf16* aRow = A + (size_t)(mBase + l15) * K;
    const __bf16* wRow[4];
    #pragma unroll
    for (int c = 0; c < 4; ++c) wRow[c] = W + (size_t)(nBase + c * 16 + l15) * K;

    // ---- stage-0 preload
    FragAB a0 = loadFragA(aRow, 0, half);
    FragAB b0[4];
    #pragma unroll
    for (int c = 0; c < 4; ++c) b0[c] = loadFragB(wRow[c], 0, half);

    // ---- pipelined main loop over K (K is a multiple of 64)
    int kb = 0;
    for (; kb + 64 < K; kb += 64) {
        FragAB a1 = loadFragA(aRow, kb + 32, half);
        FragAB b1[4];
        #pragma unroll
        for (int c = 0; c < 4; ++c) b1[c] = loadFragB(wRow[c], kb + 32, half);
        #pragma unroll
        for (int c = 0; c < 4; ++c)
            acc[c] = __builtin_amdgcn_wmma_f32_16x16x32_bf16(
                        false, a0.v, false, b0[c].v, (short)0, acc[c], false, false);

        a0 = loadFragA(aRow, kb + 64, half);
        #pragma unroll
        for (int c = 0; c < 4; ++c) b0[c] = loadFragB(wRow[c], kb + 64, half);
        #pragma unroll
        for (int c = 0; c < 4; ++c)
            acc[c] = __builtin_amdgcn_wmma_f32_16x16x32_bf16(
                        false, a1.v, false, b1[c].v, (short)0, acc[c], false, false);
    }
    // ---- tail (last 64 of K)
    {
        FragAB a1 = loadFragA(aRow, K - 32, half);
        FragAB b1[4];
        #pragma unroll
        for (int c = 0; c < 4; ++c) b1[c] = loadFragB(wRow[c], K - 32, half);
        #pragma unroll
        for (int c = 0; c < 4; ++c)
            acc[c] = __builtin_amdgcn_wmma_f32_16x16x32_bf16(
                        false, a0.v, false, b0[c].v, (short)0, acc[c], false, false);
        #pragma unroll
        for (int c = 0; c < 4; ++c)
            acc[c] = __builtin_amdgcn_wmma_f32_16x16x32_bf16(
                        false, a1.v, false, b1[c].v, (short)0, acc[c], false, false);
    }

    // epilogue: D layout VGPR p -> M = mBase + p + 8*half ; N = nBase + c*16 + l15
    #pragma unroll
    for (int c = 0; c < 4; ++c) {
        const int n = nBase + c * 16 + l15;
        const float bv = bias[n];
        #pragma unroll
        for (int p = 0; p < 8; ++p) {
            const int m = mBase + p + 8 * half;
            C[(size_t)m * N + n] = acc[c][p] + bv;
        }
    }
}

// ---------------- small init (zero h double-buffer + barrier counters) ----------------
__global__ void init_rec_state(float* hbuf, int* barCnt, int* barGen) {
    int id = blockIdx.x * 256 + threadIdx.x;
    if (id < 2 * HH) hbuf[id] = 0.0f;
    if (id == 0) { *barCnt = 0; *barGen = 0; }
}

// ---------------- persistent GRU recurrence ----------------
// Gi: (S, 3072) f32 precomputed x-path gates (r,z,n blocks of 1024), bias bi included.
// Wh: (3072, 1024) bf16 row-major, held REGISTER-RESIDENT (96 VGPRs/lane) across all steps.
// hbuf: 2x1024 f32 double buffer (parity s&1 = read, (s+1)&1 = write).
// 64 WGs x 256 threads; WG owns 16 hidden indices; 16 lanes/index split K into 64-chunks.
__global__ __launch_bounds__(256, 1) void gru_recurrence(const float* __restrict__ Gi,
                                                         const __bf16* __restrict__ Wh,
                                                         const float* __restrict__ bh,
                                                         float* hbuf,
                                                         __bf16* hist_bf16,      // may be null
                                                         float*  hist_f32,       // may be null
                                                         __bf16* sample_bf16,    // may be null
                                                         int samplePeriod, int sampleOffset,
                                                         int sampleStride,
                                                         int S, int* barCnt, int* barGen) {
    const int tid = threadIdx.x;
    const int g   = tid >> 4;            // group 0..15
    const int k   = tid & 15;            // lane within group (stays inside a wave half)
    const int j   = (blockIdx.x << 4) + g;   // owned hidden index

    __shared__ float sh[HH];

    // preload this lane's weight slice into registers: 3 rows x 64 bf16 = 96 VGPRs
    const __bf16* wr = Wh + ((size_t)j)          * HH + k * 64;
    const __bf16* wz = Wh + ((size_t)(HH + j))   * HH + k * 64;
    const __bf16* wn = Wh + ((size_t)(2*HH + j)) * HH + k * 64;
    uv4 wrg[8], wzg[8], wng[8];
    #pragma unroll
    for (int c = 0; c < 8; ++c) {
        wrg[c] = ((const uv4*)wr)[c];
        wzg[c] = ((const uv4*)wz)[c];
        wng[c] = ((const uv4*)wn)[c];
    }
    const float bhr = bh[j], bhz = bh[HH + j], bhn = bh[2*HH + j];

    for (int s = 0; s < S; ++s) {
        // acquire: make other WGs' h writes from last step visible, then stage h into LDS
        __builtin_amdgcn_fence(__ATOMIC_ACQUIRE, "agent");
        const float* hsrc = hbuf + ((s & 1) << 10);
        *(fv4*)(sh + tid * 4) = *(const fv4*)(hsrc + tid * 4);   // 256 threads x 16B = 4KB
        __syncthreads();

        // partial dot products over K chunk [k*64, k*64+64)
        float ar = 0.0f, az = 0.0f, an = 0.0f;
        const float* hk = sh + k * 64;
        #pragma unroll
        for (int c = 0; c < 8; ++c) {
            W8 r8, z8, n8;
            r8.q = wrg[c];
            z8.q = wzg[c];
            n8.q = wng[c];
            #pragma unroll
            for (int e = 0; e < 8; ++e) {
                const float hv = hk[c * 8 + e];
                ar = fmaf((float)r8.h[e], hv, ar);
                az = fmaf((float)z8.h[e], hv, az);
                an = fmaf((float)n8.h[e], hv, an);
            }
        }
        // reduce across the 16 lanes of the group (wave32: xor masks 1..8 stay in-half)
        #pragma unroll
        for (int m = 8; m >= 1; m >>= 1) {
            ar += __shfl_xor(ar, m);
            az += __shfl_xor(az, m);
            an += __shfl_xor(an, m);
        }

        if (k == 0) {
            const float gir = Gi[(size_t)s * G3H + j];
            const float giz = Gi[(size_t)s * G3H + HH + j];
            const float gin = Gi[(size_t)s * G3H + 2*HH + j];
            const float r = sigmoidf_(gir + ar + bhr);
            const float z = sigmoidf_(giz + az + bhz);
            const float n = tanhf(gin + r * (an + bhn));
            const float hnew = (1.0f - z) * n + z * sh[j];

            __hip_atomic_store(hbuf + (((s + 1) & 1) << 10) + j, hnew,
                               __ATOMIC_RELAXED, __HIP_MEMORY_SCOPE_AGENT);
            if (hist_f32)  hist_f32[(size_t)s * HH + j]  = hnew;
            if (hist_bf16) hist_bf16[(size_t)s * HH + j] = (__bf16)hnew;
            if (sample_bf16 && ((s % samplePeriod) == samplePeriod - 1)) {
                const int row = (s / samplePeriod) * sampleStride + sampleOffset;
                sample_bf16[(size_t)row * HH + j] = (__bf16)hnew;
            }
        }

        // grid-wide generation barrier (one gen increment per step)
        __threadfence();
        __syncthreads();
        if (tid == 0) {
            int old = __hip_atomic_fetch_add(barCnt, 1, __ATOMIC_ACQ_REL,
                                             __HIP_MEMORY_SCOPE_AGENT);
            if (old == (int)gridDim.x - 1) {
                __hip_atomic_store(barCnt, 0, __ATOMIC_RELAXED, __HIP_MEMORY_SCOPE_AGENT);
                __hip_atomic_store(barGen, s + 1, __ATOMIC_RELEASE, __HIP_MEMORY_SCOPE_AGENT);
            } else {
                while (__hip_atomic_load(barGen, __ATOMIC_ACQUIRE,
                                         __HIP_MEMORY_SCOPE_AGENT) < s + 1) {
                    __builtin_amdgcn_s_sleep(1);
                }
            }
        }
        __syncthreads();
    }
}

// ---------------- FC head: out[l] = sig( sig(h2[l]@fc1W^T+fc1b) @ fc2W^T + fc2b ) ----------------
__global__ __launch_bounds__(128) void fc_head(const float* __restrict__ h0,
                                               const float* __restrict__ h1,
                                               const float* __restrict__ fc1W,
                                               const float* __restrict__ fc1b,
                                               const float* __restrict__ fc2W,
                                               const float* __restrict__ fc2b,
                                               float* __restrict__ out) {
    __shared__ float y[2][64];
    const int t = threadIdx.x;
    if (t < 128) {
        const int l = t >> 6, o = t & 63;
        const float* h = l ? h1 : h0;
        float acc = fc1b[o];
        for (int c = 0; c < HH; ++c) acc = fmaf(fc1W[o * HH + c], h[c], acc);
        y[l][o] = sigmoidf_(acc);
    }
    __syncthreads();
    if (t < 2) {
        float acc = fc2b[0];
        #pragma unroll
        for (int o = 0; o < 64; ++o) acc = fmaf(fc2W[o], y[t][o], acc);
        out[t] = sigmoidf_(acc);
    }
}

// ---------------- host orchestration ----------------
extern "C" void kernel_launch(void* const* d_in, const int* in_sizes, int n_in,
                              void* d_out, int out_size, void* d_ws, size_t ws_size,
                              hipStream_t stream) {
    const float* x      = (const float*)d_in[0];
    const float* g1_Wi0 = (const float*)d_in[1];
    const float* g1_Wh0 = (const float*)d_in[2];
    const float* g1_bi0 = (const float*)d_in[3];
    const float* g1_bh0 = (const float*)d_in[4];
    const float* g1_Wi1 = (const float*)d_in[5];
    const float* g1_Wh1 = (const float*)d_in[6];
    const float* g1_bi1 = (const float*)d_in[7];
    const float* g1_bh1 = (const float*)d_in[8];
    const float* g2_Wi0 = (const float*)d_in[9];
    const float* g2_Wh0 = (const float*)d_in[10];
    const float* g2_bi0 = (const float*)d_in[11];
    const float* g2_bh0 = (const float*)d_in[12];
    const float* g2_Wi1 = (const float*)d_in[13];
    const float* g2_Wh1 = (const float*)d_in[14];
    const float* g2_bi1 = (const float*)d_in[15];
    const float* g2_bh1 = (const float*)d_in[16];
    const float* fc1_W  = (const float*)d_in[17];
    const float* fc1_b  = (const float*)d_in[18];
    const float* fc2_W  = (const float*)d_in[19];
    const float* fc2_b  = (const float*)d_in[20];
    float* out = (float*)d_out;

    // workspace carve-out (256B aligned)
    char* w = (char*)d_ws;
    auto carve = [&](size_t bytes) -> void* {
        void* p = (void*)w;
        w += (bytes + 255) & ~(size_t)255;
        return p;
    };
    __bf16* seqB   = (__bf16*)carve((size_t)SEQ1 * IN_D * 2);
    __bf16* wi0B   = (__bf16*)carve((size_t)G3H * IN_D * 2);
    __bf16* wh0B   = (__bf16*)carve((size_t)G3H * HH * 2);
    __bf16* wi1B   = (__bf16*)carve((size_t)G3H * HH * 2);
    __bf16* wh1B   = (__bf16*)carve((size_t)G3H * HH * 2);
    __bf16* wi20B  = (__bf16*)carve((size_t)G3H * HH * 2);
    __bf16* wh20B  = (__bf16*)carve((size_t)G3H * HH * 2);
    __bf16* wi21B  = (__bf16*)carve((size_t)G3H * HH * 2);
    __bf16* wh21B  = (__bf16*)carve((size_t)G3H * HH * 2);
    float*  Gi     = (float*) carve((size_t)SEQ1 * G3H * 4);   // reused by all 4 GEMMs
    __bf16* hs0B   = (__bf16*)carve((size_t)SEQ1 * HH * 2);
    __bf16* seq2B  = (__bf16*)carve((size_t)256 * HH * 2);
    __bf16* hs20B  = (__bf16*)carve((size_t)256 * HH * 2);
    float*  hist20 = (float*) carve((size_t)256 * HH * 4);
    float*  hist21 = (float*) carve((size_t)256 * HH * 4);
    float*  hbuf   = (float*) carve((size_t)2 * HH * 4);
    int*    barCnt = (int*)   carve(256);
    int*    barGen = (int*)   carve(256);
    (void)ws_size; (void)n_in; (void)in_sizes; (void)out_size;

    auto cvt = [&](const float* s, __bf16* d, int n) {
        cvt_bf16<<<(n + 255) / 256, 256, 0, stream>>>(s, d, n);
    };

    // 0) layout transforms
    pack_x_seq<<<(SEQ1 * IN_D) / 256, 256, 0, stream>>>(x, seqB);
    cvt(g1_Wi0, wi0B,  G3H * IN_D);
    cvt(g1_Wh0, wh0B,  G3H * HH);
    cvt(g1_Wi1, wi1B,  G3H * HH);
    cvt(g1_Wh1, wh1B,  G3H * HH);
    cvt(g2_Wi0, wi20B, G3H * HH);
    cvt(g2_Wh0, wh20B, G3H * HH);
    cvt(g2_Wi1, wi21B, G3H * HH);
    cvt(g2_Wh1, wh21B, G3H * HH);

    auto gemm = [&](const __bf16* A, const __bf16* W, const float* bias, float* C,
                    int M, int N, int K) {
        const int waves = (M >> 4) * (N >> 6);
        wmma_gemm_bias<<<waves / 8, 256, 0, stream>>>(A, W, bias, C, M, N, K);
    };
    auto rec = [&](const float* gi, const __bf16* Wh, const float* bh,
                   __bf16* histB, float* histF, __bf16* samp, int off, int S) {
        init_rec_state<<<(2 * HH + 255) / 256, 256, 0, stream>>>(hbuf, barCnt, barGen);
        gru_recurrence<<<NWG_REC, 256, 0, stream>>>(gi, Wh, bh, hbuf, histB, histF,
                                                    samp, BB, off, 2, S, barCnt, barGen);
    };

    // 1) gru1 layer 0
    gemm(seqB, wi0B, g1_bi0, Gi, SEQ1, G3H, IN_D);
    rec(Gi, wh0B, g1_bh0, hs0B, nullptr, seq2B, /*offset*/0, SEQ1);
    // 2) gru1 layer 1
    gemm(hs0B, wi1B, g1_bi1, Gi, SEQ1, G3H, HH);
    rec(Gi, wh1B, g1_bh1, nullptr, nullptr, seq2B, /*offset*/1, SEQ1);
    // 3) gru2 layer 0 (sequence = interleaved sampled states, 256 steps)
    gemm(seq2B, wi20B, g2_bi0, Gi, 256, G3H, HH);
    rec(Gi, wh20B, g2_bh0, hs20B, hist20, nullptr, 0, 256);
    // 4) gru2 layer 1
    gemm(hs20B, wi21B, g2_bi1, Gi, 256, G3H, HH);
    rec(Gi, wh21B, g2_bh1, nullptr, hist21, nullptr, 0, 256);
    // 5) head
    fc_head<<<1, 128, 0, stream>>>(hist20 + (size_t)255 * HH, hist21 + (size_t)255 * HH,
                                   fc1_W, fc1_b, fc2_W, fc2_b, out);
}
